// TwoDimensionalSSM_36833639531298
// MI455X (gfx1250) — matrix operand order repaired and checked by reading, not compiled, not run on hardware
//
#include <hip/hip_runtime.h>
#include <hip/hip_fp16.h>

typedef __attribute__((ext_vector_type(16))) _Float16 v16h;
typedef __attribute__((ext_vector_type(8)))  float    v8f;

#define LSIDE 16
#define LSEQ  256        // L = 16*16
#define DCH   256        // channels D
#define NCOLS 8192       // B*D = 32*256 columns
#define KDIM  1024       // 4 stacked 256x256 matrices along K

__device__ __forceinline__ float sig_half(float t) {
    // sigmoid(t) * 0.5
    return 0.5f / (1.0f + __expf(-t));
}

// ---------------------------------------------------------------------------
// Kernel 1: build the 256x1024 f16 A matrix in WMMA-A-fragment order.
//
// Per (d,n) scalar system, the impulse response on the monotone lattice:
//   G_h[dy][dx] = a1*G_h[dy][dx-1] + a2*G_v[dy][dx-1]   (G_h[*][0]=0 except origin)
//   G_v[dy][dx] = a3*G_h[dy-1][dx] + a4*G_v[dy-1][dx]   (G_v[0][*]=0 except origin)
//   G_h[0][0] = b1, G_v[0][0] = b2
// Sum over d (wave shuffle + LDS atomic) -> S[m][dy][dx], m = {h:n0,h:n1,v:n0,v:n1}.
// A[p][m*256+q] = S[m][ti-si][tj-sj] * 1{ti>=si, tj>=sj}, scattered to fragment order.
// ---------------------------------------------------------------------------
__global__ void build_A_kernel(const float* __restrict__ A1, const float* __restrict__ A2,
                               const float* __restrict__ A3, const float* __restrict__ A4,
                               const float* __restrict__ B1, const float* __restrict__ B2,
                               _Float16* __restrict__ Afrag) {
    __shared__ float S[4][LSIDE][LSIDE];
    const int tid  = threadIdx.x;   // 256 threads, one per channel d
    const int lane = tid & 31;

    for (int i = tid; i < 4 * LSIDE * LSIDE; i += 256) ((float*)S)[i] = 0.0f;
    __syncthreads();

    for (int n = 0; n < 2; ++n) {
        const int idx = tid * 2 + n;   // (D,N) row-major
        const float a1 = sig_half(A1[idx]), a2 = sig_half(A2[idx]);
        const float a3 = sig_half(A3[idx]), a4 = sig_half(A4[idx]);
        const float b1 = sig_half(B1[idx]), b2 = sig_half(B2[idx]);

        float ghp[LSIDE], gvp[LSIDE], gh[LSIDE], gv[LSIDE];
        for (int dy = 0; dy < LSIDE; ++dy) {
#pragma unroll
            for (int dx = 0; dx < LSIDE; ++dx)
                gv[dx] = (dy == 0) ? ((dx == 0) ? b2 : 0.0f)
                                   : (a3 * ghp[dx] + a4 * gvp[dx]);
            gh[0] = (dy == 0) ? b1 : 0.0f;
#pragma unroll
            for (int dx = 1; dx < LSIDE; ++dx)
                gh[dx] = a1 * gh[dx - 1] + a2 * gv[dx - 1];

            // reduce across the 32 lanes (32 of the 256 d's), then LDS atomic
#pragma unroll
            for (int dx = 0; dx < LSIDE; ++dx) {
                float vh = gh[dx], vv = gv[dx];
                for (int off = 16; off > 0; off >>= 1) {
                    vh += __shfl_xor(vh, off, 32);
                    vv += __shfl_xor(vv, off, 32);
                }
                if (lane == 0) {
                    atomicAdd(&S[n][dy][dx], vh);
                    atomicAdd(&S[2 + n][dy][dx], vv);
                }
            }
#pragma unroll
            for (int dx = 0; dx < LSIDE; ++dx) { ghp[dx] = gh[dx]; gvp[dx] = gv[dx]; }
        }
    }
    __syncthreads();

    // Scatter A into v_wmma_f32_16x16x32_f16 A fragments, this code's contiguous K order (B uses the same):
    // lane = M + 16*hi, hi selects K in [16,32); element e holds K = 16*hi + e.
    const int p  = tid;
    const int r  = p >> 4;          // row tile
    const int M  = p & 15;
    const int ti = p >> 4, tj = p & 15;
    for (int k = 0; k < KDIM; ++k) {
        const int m = k >> 8, q = k & 255;
        const int si = q >> 4, sj = q & 15;
        float val = 0.0f;
        if (ti >= si && tj >= sj) val = S[m][ti - si][tj - sj];
        const int kk = k >> 5;      // K-step (32 per WMMA)
        const int K  = k & 31;
        const int hi = K >> 4;
        const int ln = M + (hi << 4);
        const int base = hi << 4;
        const int e = K - base;
        Afrag[(((r * 32 + kk) * 32 + ln) << 4) + e] = (_Float16)val;
    }
}

// ---------------------------------------------------------------------------
// Kernel 2: pack B^T in f16: Bt[col*1024 + k] = coef_m[d] * x[q, col]
// where k = m*256+q, col = b*256+d, coef = {C1[:,0],C1[:,1],C2[:,0],C2[:,1]}.
// K contiguous per column -> each WMMA B-fragment lane load is one 32B v16h.
// ---------------------------------------------------------------------------
__global__ void pack_B_kernel(const float* __restrict__ x,
                              const float* __restrict__ C1,
                              const float* __restrict__ C2,
                              _Float16* __restrict__ Bt) {
    const int g   = blockIdx.x * blockDim.x + threadIdx.x;  // 8192 cols * 128 groups
    const int col = g >> 7;
    const int kg  = g & 127;
    const int d   = col & 255;

    union { _Float16 h[8]; uint4 u; } vals;
    const int k0 = kg << 3;
    const int m  = k0 >> 8;          // constant across the 8 k's
    const int n  = m & 1;
    const float coef = (m < 2) ? C1[d * 2 + n] : C2[d * 2 + n];
#pragma unroll
    for (int j = 0; j < 8; ++j) {
        const int q = (k0 + j) & 255;
        vals.h[j] = (_Float16)(coef * x[q * NCOLS + col]);
    }
    *(uint4*)(Bt + col * KDIM + k0) = vals.u;
}

// ---------------------------------------------------------------------------
// Kernel 3: C(256x8192) = A(256x1024) x B(1024x8192) via v_wmma_f32_16x16x32_f16.
// One wave owns a 16x64 tile (4 accumulators, A fragment reused 4x), 32 K-steps.
// Epilogue: silu(acc + x*omega).
// ---------------------------------------------------------------------------
__global__ void ssm_gemm_kernel(const _Float16* __restrict__ Afrag,
                                const _Float16* __restrict__ Bt,
                                const float* __restrict__ x,
                                const float* __restrict__ omega,
                                float* __restrict__ out) {
    const int wave = (blockIdx.x * blockDim.x + threadIdx.x) >> 5;  // 0..2047
    const int lane = threadIdx.x & 31;
    const int r  = wave >> 7;   // 0..15 row tile of 16
    const int cg = wave & 127;  // 0..127 col group of 64

    v8f acc0 = {}, acc1 = {}, acc2 = {}, acc3 = {};

    // B-operand layout: N = lane&15; lanes 0-15 hold K=0..15, lanes 16-31 K=16..31
    const int nlo = lane & 15;
    const int khi = (lane >> 4) << 4;
    const _Float16* bbase = Bt + (size_t)(cg * 64 + nlo) * KDIM + khi;
    const _Float16* abase = Afrag + ((size_t)(r * 32) * 32 + lane) * 16;

    for (int kk = 0; kk < 32; ++kk) {
        v16h a  = *(const v16h*)(abase + (size_t)kk * 32 * 16);
        v16h b0 = *(const v16h*)(bbase + kk * 32);
        v16h b1 = *(const v16h*)(bbase + 16 * KDIM + kk * 32);
        v16h b2 = *(const v16h*)(bbase + 32 * KDIM + kk * 32);
        v16h b3 = *(const v16h*)(bbase + 48 * KDIM + kk * 32);
        acc0 = __builtin_amdgcn_wmma_f32_16x16x32_f16(false, a, false, b0, (short)0, acc0, false, false);
        acc1 = __builtin_amdgcn_wmma_f32_16x16x32_f16(false, a, false, b1, (short)0, acc1, false, false);
        acc2 = __builtin_amdgcn_wmma_f32_16x16x32_f16(false, a, false, b2, (short)0, acc2, false, false);
        acc3 = __builtin_amdgcn_wmma_f32_16x16x32_f16(false, a, false, b3, (short)0, acc3, false, false);
    }

    // C/D layout: element i -> M = i + 8*(lane>=16), N = lane&15
    const int Mbase = (lane >> 4) << 3;
    const int N = lane & 15;
    v8f accs[4] = {acc0, acc1, acc2, acc3};
#pragma unroll
    for (int t = 0; t < 4; ++t) {
        const int col = cg * 64 + t * 16 + N;
        const float om = omega[col & 255];
#pragma unroll
        for (int i = 0; i < 8; ++i) {
            const int p = r * 16 + Mbase + i;
            const float s = accs[t][i] + x[p * NCOLS + col] * om;
            out[p * NCOLS + col] = s / (1.0f + __expf(-s));   // silu
        }
    }
}

extern "C" void kernel_launch(void* const* d_in, const int* in_sizes, int n_in,
                              void* d_out, int out_size, void* d_ws, size_t ws_size,
                              hipStream_t stream) {
    const float* x     = (const float*)d_in[0];
    const float* A1    = (const float*)d_in[1];
    const float* A2    = (const float*)d_in[2];
    const float* A3    = (const float*)d_in[3];
    const float* A4    = (const float*)d_in[4];
    const float* B1    = (const float*)d_in[5];
    const float* B2    = (const float*)d_in[6];
    const float* C1    = (const float*)d_in[7];
    const float* C2    = (const float*)d_in[8];
    const float* omega = (const float*)d_in[9];
    float* out = (float*)d_out;

    // workspace: [ Afrag: 256K halves = 512KB ][ Bt: 1024*8192 halves = 16MB ]
    _Float16* Afrag = (_Float16*)d_ws;
    _Float16* Bt    = (_Float16*)((char*)d_ws + (512u * 1024u));

    build_A_kernel<<<1, 256, 0, stream>>>(A1, A2, A3, A4, B1, B2, Afrag);
    pack_B_kernel<<<(NCOLS * 128) / 256, 256, 0, stream>>>(x, C1, C2, Bt);
    ssm_gemm_kernel<<<(2048 * 32) / 256, 256, 0, stream>>>(Afrag, Bt, x, omega, out);
}
